// SlicedLSTM_82214263980094
// MI455X (gfx1250) — compile-verified, matrix operand order checked by reading, not run on hardware
//
#include <hip/hip_runtime.h>
#include <hip/hip_bf16.h>

// ---------------------------------------------------------------------------
// SlicedLSTM for MI455X (gfx1250, wave32, WMMA + async-to-LDS).
//  - bf16 WMMA (v_wmma_f32_16x16x32_bf16) for both GEMM stages, f32 accum.
//  - Weights transposed once to bf16 [N][K] so all fragment loads are b128.
//  - 4-wave workgroups: B (weight) tile staged to LDS with
//    global_load_async_to_lds_b128 (ASYNCcnt), double-buffered; all 4 waves
//    share the tile via ds_load_b128 reads matching the WMMA B striping.
//  - Each wave register-blocks 4 N-tiles (A-fragment reused across 4 WMMAs).
//  - Recurrence = 3 kernels per timestep on `stream` (graph-captured).
// ---------------------------------------------------------------------------

typedef __bf16 bf16_t;
typedef __attribute__((ext_vector_type(16))) __bf16 v16bf;
typedef __attribute__((ext_vector_type(8)))  __bf16 v8bf;
typedef __attribute__((ext_vector_type(8)))  float  v8f;

#define NSLICE 4
#define IN_SZ  128
#define HID_SZ 256
#define HTOT   1024      // NSLICE*HID_SZ
#define FTOT   512       // NSLICE*IN_SZ
#define BSZ    64
#define TSZ    512
#define KA     384       // IN_SZ + HID_SZ (fused [x|h] GEMM K)

// ---- workspace layout (bytes) --------------------------------------------
static const size_t WS_WUT = 0;                                           // bf16 [S][1024][384]
static const size_t WS_WCT = WS_WUT + (size_t)NSLICE * HTOT * KA * 2;     // bf16 [4096][1024]
static const size_t WS_HBF = WS_WCT + (size_t)4 * HTOT * HTOT * 2;        // bf16 [B][H]
static const size_t WS_CST = WS_HBF + (size_t)BSZ * HTOT * 2;             // f32  [B][H]
static const size_t WS_TOT = WS_CST + (size_t)BSZ * HTOT * 4;             // bf16 [4][B][H]
static const size_t WS_GAC = WS_TOT + (size_t)4 * BSZ * HTOT * 2;         // f32  [4][B][H]
// total ~13.6 MB

__device__ __forceinline__ float sigmoidf_(float x) {
    return 1.0f / (1.0f + __expf(-x));
}
__device__ __forceinline__ v8bf load8_bf(const bf16_t* p) {
    return *reinterpret_cast<const v8bf*>(p);   // global_load_b128
}
__device__ __forceinline__ v8bf cvt8_f32(const float* p) {
    v8f f = *reinterpret_cast<const v8f*>(p);
    v8bf r;
#pragma unroll
    for (int i = 0; i < 8; ++i) r[i] = (bf16_t)f[i];
    return r;
}

union FragU { v16bf v; v8bf h[2]; };

// --- CDNA5 async copy: global -> LDS (16B per lane), tracked by ASYNCcnt ---
__device__ __forceinline__ void async_copy16(const bf16_t* gsrc, bf16_t* ldst) {
    unsigned loff = (unsigned)(uintptr_t)(__attribute__((address_space(3))) bf16_t*)ldst;
    unsigned long long gaddr = (unsigned long long)(uintptr_t)gsrc;
    asm volatile("global_load_async_to_lds_b128 %0, %1, off"
                 :: "v"(loff), "v"(gaddr) : "memory");
}
__device__ __forceinline__ void wait_async0() {
    asm volatile("s_wait_asynccnt 0x0" ::: "memory");
}

// Stage a 64-row x 32-col bf16 tile (4 KB) from row-major [*, LDK] source into
// LDS. 128 threads x 2 chunks of 16B each. src points at (row0, col kb).
template <int LDK>
__device__ __forceinline__ void stage_tile(const bf16_t* __restrict__ src,
                                           bf16_t* __restrict__ dst, int tid) {
#pragma unroll
    for (int i = 0; i < 2; ++i) {
        const int c    = tid + i * 128;      // 0..255
        const int row  = c >> 2;
        const int part = c & 3;
        async_copy16(src + (size_t)row * LDK + part * 8, dst + row * 32 + part * 8);
    }
}

// ---------------------------------------------------------------------------
// One-time prep kernels
// ---------------------------------------------------------------------------
__global__ void prep_wu_kernel(const float* __restrict__ Ws,
                               const float* __restrict__ Us,
                               bf16_t* __restrict__ WUt) {
    size_t tid = (size_t)blockIdx.x * blockDim.x + threadIdx.x;
    if (tid >= (size_t)NSLICE * HTOT * KA) return;
    int k = (int)(tid % KA);
    int n = (int)((tid / KA) % HTOT);
    int s = (int)(tid / ((size_t)KA * HTOT));
    float v = (k < IN_SZ)
        ? Ws[((size_t)s * IN_SZ  + k)           * HTOT + n]
        : Us[((size_t)s * HID_SZ + (k - IN_SZ)) * HTOT + n];
    WUt[tid] = (bf16_t)v;
}

__global__ void prep_wc_kernel(const float* __restrict__ Wc,
                               bf16_t* __restrict__ Wct) {
    size_t tid = (size_t)blockIdx.x * blockDim.x + threadIdx.x;
    if (tid >= (size_t)4 * HTOT * HTOT) return;
    int k = (int)(tid & (HTOT - 1));
    int n = (int)(tid >> 10);
    Wct[tid] = (bf16_t)Wc[(size_t)k * (4 * HTOT) + n];
}

__global__ void init_state_kernel(bf16_t* __restrict__ hbf, float* __restrict__ cst) {
    int idx = blockIdx.x * blockDim.x + threadIdx.x;
    if (idx >= BSZ * HTOT) return;
    hbf[idx] = (bf16_t)0.0f;
    cst[idx] = 0.0f;
}

// ---------------------------------------------------------------------------
// Stage 1: slice gates.  [x_s(t)|h_s] (64x384) @ WU_s^T-packed (1024x384)
// grid = (HTOT/64=16, 1, s=4), block = 128 (4 waves).
// Wave w computes rows w*16..w*16+15, 4 N-tiles of 16 (64 cols/block).
// ---------------------------------------------------------------------------
__global__ __launch_bounds__(128)
void gates_kernel(const float* __restrict__ x,
                  const float* __restrict__ biases,
                  const bf16_t* __restrict__ WUt,
                  const bf16_t* __restrict__ hbf,
                  bf16_t* __restrict__ totals,
                  int t) {
    __shared__ alignas(16) bf16_t tileB[2][64 * 32];   // 8 KB double buffer

    const int tid  = threadIdx.x;
    const int lane = tid & 31;
    const int w    = tid >> 5;
    const int l16  = lane & 15;
    const int half = lane >> 4;
    const int s    = blockIdx.z;
    const int n0   = blockIdx.x * 64;
    const int brow = w * 16 + l16;

    const float*  xrow  = x   + ((size_t)brow * TSZ + t) * FTOT + (size_t)s * IN_SZ;
    const bf16_t* hrow  = hbf + (size_t)brow * HTOT + (size_t)s * HID_SZ;
    const bf16_t* wbase = WUt + ((size_t)s * HTOT + n0) * KA;

    v8f acc[4] = {};
    stage_tile<KA>(wbase, &tileB[0][0], tid);

    for (int kb = 0, buf = 0; kb < KA; kb += 32, buf ^= 1) {
        wait_async0();
        __syncthreads();
        if (kb + 32 < KA) stage_tile<KA>(wbase + (kb + 32), &tileB[buf ^ 1][0], tid);

        // A fragment: region (x vs h) chosen by uniform (kb + c*16) < 128.
        FragU A;
#pragma unroll
        for (int c = 0; c < 2; ++c) {
            const int kc = kb + c * 16;
            const int k0 = kc + half * 8;
            if (kc < IN_SZ) A.h[c] = cvt8_f32(xrow + k0);
            else            A.h[c] = load8_bf(hrow + (k0 - IN_SZ));
        }
#pragma unroll
        for (int nt = 0; nt < 4; ++nt) {
            const v8bf* lrow = (const v8bf*)&tileB[buf][(nt * 16 + l16) * 32];
            FragU Bm;
            Bm.h[0] = lrow[half];        // K = half*8 .. +7
            Bm.h[1] = lrow[2 + half];    // K = 16 + half*8 .. +7
            acc[nt] = __builtin_amdgcn_wmma_f32_16x16x32_bf16(
                          false, A.v, false, Bm.v, (short)0, acc[nt], false, false);
        }
        __syncthreads();
    }

    // Epilogue: bias + activation, scatter into bf16 totals[q][b][H].
    // 64-col block lies in one quarter of 4*HID, so q is uniform.
    const int q   = n0 >> 8;                           // 0:i 1:f 2:g 3:o
    const int jj0 = s * HID_SZ + (n0 & (HID_SZ - 1));
    bf16_t* tq = totals + (size_t)q * (BSZ * HTOT);
#pragma unroll
    for (int nt = 0; nt < 4; ++nt) {
        const int   nn   = nt * 16 + l16;
        const float bias = biases[s * (4 * HID_SZ) + n0 + nn];
#pragma unroll
        for (int r = 0; r < 8; ++r) {
            const int bb = w * 16 + r + half * 8;      // C layout: VGPR r -> row
            const float v = acc[nt][r] + bias;
            const float a = (q == 2) ? tanhf(v) : sigmoidf_(v);
            tq[(size_t)bb * HTOT + jj0 + nn] = (bf16_t)a;
        }
    }
}

// ---------------------------------------------------------------------------
// Stage 2: connector.  totals[q] (64x1024) @ Wc_q^T-packed (1024x1024)
// grid = (HTOT/64=16, 1, q=4), block = 128 (4 waves).
// ---------------------------------------------------------------------------
__global__ __launch_bounds__(128)
void connector_kernel(const bf16_t* __restrict__ totals,
                      const bf16_t* __restrict__ Wct,
                      const float* __restrict__ cb,
                      float* __restrict__ gact) {
    __shared__ alignas(16) bf16_t tileB[2][64 * 32];

    const int tid  = threadIdx.x;
    const int lane = tid & 31;
    const int w    = tid >> 5;
    const int l16  = lane & 15;
    const int half = lane >> 4;
    const int q    = blockIdx.z;
    const int n0   = blockIdx.x * 64;
    const int brow = w * 16 + l16;

    const bf16_t* arow  = totals + (size_t)q * (BSZ * HTOT) + (size_t)brow * HTOT;
    const bf16_t* wbase = Wct + ((size_t)q * HTOT + n0) * HTOT;

    v8f acc[4] = {};
    stage_tile<HTOT>(wbase, &tileB[0][0], tid);

    for (int kb = 0, buf = 0; kb < HTOT; kb += 32, buf ^= 1) {
        wait_async0();
        __syncthreads();
        if (kb + 32 < HTOT) stage_tile<HTOT>(wbase + (kb + 32), &tileB[buf ^ 1][0], tid);

        FragU A;
        A.h[0] = load8_bf(arow + kb + half * 8);
        A.h[1] = load8_bf(arow + kb + 16 + half * 8);
#pragma unroll
        for (int nt = 0; nt < 4; ++nt) {
            const v8bf* lrow = (const v8bf*)&tileB[buf][(nt * 16 + l16) * 32];
            FragU Bm;
            Bm.h[0] = lrow[half];
            Bm.h[1] = lrow[2 + half];
            acc[nt] = __builtin_amdgcn_wmma_f32_16x16x32_bf16(
                          false, A.v, false, Bm.v, (short)0, acc[nt], false, false);
        }
        __syncthreads();
    }

    float* gq = gact + (size_t)q * (BSZ * HTOT);
#pragma unroll
    for (int nt = 0; nt < 4; ++nt) {
        const int   nn   = nt * 16 + l16;
        const float bias = cb[q * HTOT + n0 + nn];
#pragma unroll
        for (int r = 0; r < 8; ++r) {
            const int bb = w * 16 + r + half * 8;
            const float v = acc[nt][r] + bias;
            gq[(size_t)bb * HTOT + n0 + nn] = (q == 2) ? tanhf(v) : sigmoidf_(v);
        }
    }
}

// ---------------------------------------------------------------------------
// Stage 3: cell/hidden update + outputs.
// ---------------------------------------------------------------------------
__global__ void cell_update_kernel(const float* __restrict__ gact,
                                   float* __restrict__ cst,
                                   bf16_t* __restrict__ hbf,
                                   float* __restrict__ seq_out,
                                   float* __restrict__ out_h,
                                   float* __restrict__ out_c,
                                   int t, int last) {
    const int idx = blockIdx.x * blockDim.x + threadIdx.x;
    if (idx >= BSZ * HTOT) return;
    const int b = idx >> 10, n = idx & (HTOT - 1);
    const float i = gact[0 * (BSZ * HTOT) + idx];
    const float f = gact[1 * (BSZ * HTOT) + idx];
    const float g = gact[2 * (BSZ * HTOT) + idx];
    const float o = gact[3 * (BSZ * HTOT) + idx];
    const float cn = f * cst[idx] + i * g;
    const float h  = o * tanhf(cn);
    cst[idx] = cn;
    hbf[idx] = (bf16_t)h;
    seq_out[((size_t)b * TSZ + t) * HTOT + n] = h;
    if (last) { out_h[idx] = h; out_c[idx] = cn; }
}

// ---------------------------------------------------------------------------
extern "C" void kernel_launch(void* const* d_in, const int* in_sizes, int n_in,
                              void* d_out, int out_size, void* d_ws, size_t ws_size,
                              hipStream_t stream) {
    const float* x   = (const float*)d_in[0];   // [B,T,F]
    const float* Ws  = (const float*)d_in[1];   // [S,128,1024]
    const float* Us  = (const float*)d_in[2];   // [S,256,1024]
    const float* bia = (const float*)d_in[3];   // [S,1024]
    const float* Wc  = (const float*)d_in[4];   // [1024,4096]
    const float* cb  = (const float*)d_in[5];   // [4096]

    char* ws = (char*)d_ws;
    bf16_t* WUt = (bf16_t*)(ws + WS_WUT);
    bf16_t* Wct = (bf16_t*)(ws + WS_WCT);
    bf16_t* hbf = (bf16_t*)(ws + WS_HBF);
    float*  cst = (float*) (ws + WS_CST);
    bf16_t* tot = (bf16_t*)(ws + WS_TOT);
    float*  gac = (float*) (ws + WS_GAC);

    float* seq_out = (float*)d_out;                            // [B,T,H]
    float* out_h   = seq_out + (size_t)BSZ * TSZ * HTOT;       // [B,H]
    float* out_c   = out_h + (size_t)BSZ * HTOT;               // [B,H]

    {
        const size_t nwu = (size_t)NSLICE * HTOT * KA;
        prep_wu_kernel<<<dim3((nwu + 255) / 256), dim3(256), 0, stream>>>(Ws, Us, WUt);
        const size_t nwc = (size_t)4 * HTOT * HTOT;
        prep_wc_kernel<<<dim3((nwc + 255) / 256), dim3(256), 0, stream>>>(Wc, Wct);
        init_state_kernel<<<dim3((BSZ * HTOT + 255) / 256), dim3(256), 0, stream>>>(hbf, cst);
    }

    for (int t = 0; t < TSZ; ++t) {
        gates_kernel<<<dim3(HTOT / 64, 1, NSLICE), dim3(128), 0, stream>>>(
            x, bia, WUt, hbf, tot, t);
        connector_kernel<<<dim3(HTOT / 64, 1, 4), dim3(128), 0, stream>>>(
            tot, Wct, cb, gac);
        cell_update_kernel<<<dim3((BSZ * HTOT + 255) / 256), dim3(256), 0, stream>>>(
            gac, cst, hbf, seq_out, out_h, out_c, t, (t == TSZ - 1) ? 1 : 0);
    }
}
